// PointNet2FeaturePropagator_53506702574032
// MI455X (gfx1250) — matrix operand order repaired and checked by reading, not compiled, not run on hardware
//
#include <hip/hip_runtime.h>
#include <cstdint>
#include <cstddef>

#define BB   8
#define NN   8192
#define MM   2048
#define CC   256
#define CPC  256
#define D1C  256
#define D2C  256
#define KIN  512      // CP + C
#define BN_EPS 1e-5f

// GEMM tiling
#define TM 64         // block tile M (channels)
#define TN 128        // block tile N (points)
#define TK 32         // K step (one bf16 WMMA depth)

typedef __attribute__((ext_vector_type(16))) __bf16 v16bf;
typedef __attribute__((ext_vector_type(8)))  float  v8f;

union Frag {
    uint4 q[2];
    v16bf v;
};

__device__ __forceinline__ unsigned short f32_to_bf16(float f) {
    unsigned int u = __float_as_uint(f);
    u += 0x7FFFu + ((u >> 16) & 1u);          // round-to-nearest-even
    return (unsigned short)(u >> 16);
}

// ---------------------------------------------------------------------------
// 1) 3-NN search + inverse-distance weights.
// ---------------------------------------------------------------------------
__global__ __launch_bounds__(256) void knn_kernel(
    const float* __restrict__ xyz, const float* __restrict__ xyz_prev,
    int* __restrict__ idx_out, float* __restrict__ w_out) {
    __shared__ float lq[512 * 3];
    int b = blockIdx.y;
    int n = blockIdx.x * 256 + threadIdx.x;
    float px = xyz[((size_t)b * NN + n) * 3 + 0];
    float py = xyz[((size_t)b * NN + n) * 3 + 1];
    float pz = xyz[((size_t)b * NN + n) * 3 + 2];
    float b0 = 3.4e38f, b1 = 3.4e38f, b2 = 3.4e38f;
    int   j0 = 0, j1 = 0, j2 = 0;
    for (int m0 = 0; m0 < MM; m0 += 512) {
        __syncthreads();
        for (int i = threadIdx.x; i < 512 * 3; i += 256)
            lq[i] = xyz_prev[(size_t)b * MM * 3 + (size_t)m0 * 3 + i];
        __syncthreads();
        for (int j = 0; j < 512; ++j) {
            float dx = px - lq[j * 3 + 0];
            float dy = py - lq[j * 3 + 1];
            float dz = pz - lq[j * 3 + 2];
            float d  = dx * dx + dy * dy + dz * dz;
            int  mg  = m0 + j;
            if (d < b0)      { b2 = b1; j2 = j1; b1 = b0; j1 = j0; b0 = d; j0 = mg; }
            else if (d < b1) { b2 = b1; j2 = j1; b1 = d;  j1 = mg; }
            else if (d < b2) { b2 = d;  j2 = mg; }
        }
    }
    float d0 = sqrtf(fmaxf(b0, 0.f));
    float d1 = sqrtf(fmaxf(b1, 0.f));
    float d2 = sqrtf(fmaxf(b2, 0.f));
    float i0 = 1.f / (d0 + 1e-8f), i1 = 1.f / (d1 + 1e-8f), i2 = 1.f / (d2 + 1e-8f);
    float s  = 1.f / (i0 + i1 + i2);
    size_t o = ((size_t)b * NN + n) * 3;
    idx_out[o + 0] = j0; idx_out[o + 1] = j1; idx_out[o + 2] = j2;
    w_out[o + 0] = i0 * s; w_out[o + 1] = i1 * s; w_out[o + 2] = i2 * s;
}

// ---------------------------------------------------------------------------
// 2) Weight conversion f32 -> bf16.
// ---------------------------------------------------------------------------
__global__ __launch_bounds__(256) void convert_w_kernel(
    const float* __restrict__ w1, const float* __restrict__ w2,
    unsigned short* __restrict__ w1b, unsigned short* __restrict__ w2b) {
    int i = blockIdx.x * 256 + threadIdx.x;
    if (i < D1C * KIN) {
        w1b[i] = f32_to_bf16(w1[i]);
    } else {
        int j = i - D1C * KIN;
        if (j < D2C * D1C) w2b[j] = f32_to_bf16(w2[j]);
    }
}

// ---------------------------------------------------------------------------
// 3) Interpolate + concat -> x bf16 [B][KIN][NN].
// ---------------------------------------------------------------------------
__global__ __launch_bounds__(256) void interp_concat_kernel(
    const float* __restrict__ feats, const float* __restrict__ feats_prev,
    const int* __restrict__ idx, const float* __restrict__ wgt,
    unsigned short* __restrict__ xb) {
    int c = blockIdx.x;   // 0..KIN-1
    int b = blockIdx.y;
    unsigned short* dst = xb + ((size_t)b * KIN + c) * NN;
    if (c < CPC) {
        __shared__ float row[MM];
        for (int i = threadIdx.x; i < MM; i += 256)
            row[i] = feats_prev[((size_t)b * CPC + c) * MM + i];
        __syncthreads();
        for (int n = threadIdx.x; n < NN; n += 256) {
            size_t o = ((size_t)b * NN + n) * 3;
            float v = wgt[o + 0] * row[idx[o + 0]] +
                      wgt[o + 1] * row[idx[o + 1]] +
                      wgt[o + 2] * row[idx[o + 2]];
            dst[n] = f32_to_bf16(v);
        }
    } else {
        const float* src = feats + ((size_t)b * CC + (c - CPC)) * NN;
        for (int n = threadIdx.x; n < NN; n += 256)
            dst[n] = f32_to_bf16(src[n]);
    }
}

// ---------------------------------------------------------------------------
// 4) Zero the BN-stats accumulators (1024 floats).
// ---------------------------------------------------------------------------
__global__ __launch_bounds__(256) void zero_kernel(float* __restrict__ p) {
    p[blockIdx.x * 256 + threadIdx.x] = 0.f;
}

// ---------------------------------------------------------------------------
// 5) GEMM1: Y1 = w1 * x + b1, bf16 WMMA, double-buffered LDS, register-blocked
//    16x64 per wave.  All four B-fragments loaded into independent registers
//    before the WMMA clause so one DScnt wait covers the whole group.
// ---------------------------------------------------------------------------
__global__ __launch_bounds__(256) void gemm1_kernel(
    const unsigned short* __restrict__ A,   // [D1][KIN] bf16
    const unsigned short* __restrict__ Bx,  // [B][KIN][NN] bf16
    const float* __restrict__ bias,         // [D1]
    float* __restrict__ Y,                  // [B][D1][NN]
    float* __restrict__ gsum, float* __restrict__ gsq) {
    __shared__ __align__(16) unsigned short ldsB[2][TN][40];
    __shared__ float bsum[TM], bsq[TM];
    int bz = blockIdx.z;
    int m0 = blockIdx.y * TM;
    int n0 = blockIdx.x * TN;
    int lane = threadIdx.x & 31;
    int wv   = threadIdx.x >> 5;
    int mi = wv & 3;                        // 4 M-subtiles
    int ni = wv >> 2;                       // 2 N-groups of 64
    int koffA = (lane >> 4) * 16;           // A frag: lanes<16 K{0..15}
    int mA    = m0 + mi * 16 + (lane & 15);
    int nlb   = ni * 64 + (lane & 15);      // base N row for B frags
    int koffB = (lane >> 4) * 16;
    if (threadIdx.x < TM) { bsum[threadIdx.x] = 0.f; bsq[threadIdx.x] = 0.f; }
    v8f acc[4] = {};
    int tk = threadIdx.x >> 3;              // 0..31 (K within tile)
    int tn = (threadIdx.x & 7) * 16;        // 0..112 (N within tile)
    const unsigned short* bsrc = Bx + ((size_t)bz * KIN + tk) * NN + n0 + tn;

    auto stage = [&](int k0, int buf) {
        const unsigned short* s = bsrc + (size_t)k0 * NN;
        uint4 q0 = *(const uint4*)(s);
        uint4 q1 = *(const uint4*)(s + 8);
        unsigned int w_[8] = {q0.x, q0.y, q0.z, q0.w, q1.x, q1.y, q1.z, q1.w};
#pragma unroll
        for (int i = 0; i < 8; ++i) {
            ldsB[buf][tn + 2 * i    ][tk] = (unsigned short)(w_[i] & 0xffffu);
            ldsB[buf][tn + 2 * i + 1][tk] = (unsigned short)(w_[i] >> 16);
        }
    };
    auto compute = [&](int k0, int buf) {
        // Issue A (global) first, then all 8 B ds_loads, then 4 WMMAs.
        Frag af;
        const unsigned short* arow = A + (size_t)mA * KIN + k0 + koffA;
        af.q[0] = *(const uint4*)(arow);
        af.q[1] = *(const uint4*)(arow + 8);
        Frag bf0, bf1, bf2, bf3;
        const unsigned short* br0 = &ldsB[buf][nlb +  0][koffB];
        const unsigned short* br1 = &ldsB[buf][nlb + 16][koffB];
        const unsigned short* br2 = &ldsB[buf][nlb + 32][koffB];
        const unsigned short* br3 = &ldsB[buf][nlb + 48][koffB];
        bf0.q[0] = *(const uint4*)(br0); bf0.q[1] = *(const uint4*)(br0 + 8);
        bf1.q[0] = *(const uint4*)(br1); bf1.q[1] = *(const uint4*)(br1 + 8);
        bf2.q[0] = *(const uint4*)(br2); bf2.q[1] = *(const uint4*)(br2 + 8);
        bf3.q[0] = *(const uint4*)(br3); bf3.q[1] = *(const uint4*)(br3 + 8);
        acc[0] = __builtin_amdgcn_wmma_f32_16x16x32_bf16(
            false, af.v, false, bf0.v, (short)0, acc[0], false, false);
        acc[1] = __builtin_amdgcn_wmma_f32_16x16x32_bf16(
            false, af.v, false, bf1.v, (short)0, acc[1], false, false);
        acc[2] = __builtin_amdgcn_wmma_f32_16x16x32_bf16(
            false, af.v, false, bf2.v, (short)0, acc[2], false, false);
        acc[3] = __builtin_amdgcn_wmma_f32_16x16x32_bf16(
            false, af.v, false, bf3.v, (short)0, acc[3], false, false);
    };

    const int nsteps = KIN / TK;            // 16
    stage(0, 0);
    for (int t = 0; t < nsteps; ++t) {
        __syncthreads();
        if (t + 1 < nsteps) stage((t + 1) * TK, (t + 1) & 1);
        compute(t * TK, t & 1);
    }

    // Epilogue: write Y, fold per-channel partial sums.
#pragma unroll
    for (int j = 0; j < 4; ++j) {
#pragma unroll
        for (int r = 0; r < 8; ++r) {
            int mloc = mi * 16 + r + ((lane >> 4) << 3);
            int m = m0 + mloc;
            int n = n0 + ni * 64 + j * 16 + (lane & 15);
            float val = acc[j][r] + bias[m];
            Y[((size_t)bz * D1C + m) * NN + n] = val;
            float s = val, q = val * val;
#pragma unroll
            for (int off = 1; off < 16; off <<= 1) {
                s += __shfl_xor(s, off, 16);
                q += __shfl_xor(q, off, 16);
            }
            if ((lane & 15) == 0) {
                atomicAdd(&bsum[mloc], s);
                atomicAdd(&bsq[mloc], q);
            }
        }
    }
    __syncthreads();
    if (threadIdx.x < TM) {
        atomicAdd(&gsum[m0 + threadIdx.x], bsum[threadIdx.x]);
        atomicAdd(&gsq[m0 + threadIdx.x],  bsq[threadIdx.x]);
    }
}

// ---------------------------------------------------------------------------
// 6) Finalize BN stats -> folded scale/shift (1 block, 256 threads).
// ---------------------------------------------------------------------------
__global__ __launch_bounds__(256) void finalize_kernel(
    const float* __restrict__ gsum, const float* __restrict__ gsq,
    const float* __restrict__ g, const float* __restrict__ beta,
    float* __restrict__ scale, float* __restrict__ shift) {
    int c = threadIdx.x;
    float cnt  = (float)(BB * NN);
    float mean = gsum[c] / cnt;
    float var  = gsq[c] / cnt - mean * mean;
    float sc   = g[c] * rsqrtf(var + BN_EPS);
    scale[c] = sc;
    shift[c] = beta[c] - mean * sc;
}

// ---------------------------------------------------------------------------
// 7) GEMM2: BN1+ReLU fused into B-tile staging; same blocking; fused stats2.
// ---------------------------------------------------------------------------
__global__ __launch_bounds__(256) void gemm2_kernel(
    const unsigned short* __restrict__ A,   // [D2][D1] bf16
    const float* __restrict__ Y1,           // [B][D1][NN] pre-BN layer-1 out
    const float* __restrict__ scale1, const float* __restrict__ shift1,
    const float* __restrict__ bias,         // [D2]
    float* __restrict__ Y,                  // [B][D2][NN]  (= d_out)
    float* __restrict__ gsum, float* __restrict__ gsq) {
    __shared__ __align__(16) unsigned short ldsB[2][TN][40];
    __shared__ float bsum[TM], bsq[TM];
    int bz = blockIdx.z;
    int m0 = blockIdx.y * TM;
    int n0 = blockIdx.x * TN;
    int lane = threadIdx.x & 31;
    int wv   = threadIdx.x >> 5;
    int mi = wv & 3;
    int ni = wv >> 2;
    int koffA = (lane >> 4) * 16;
    int mA    = m0 + mi * 16 + (lane & 15);
    int nlb   = ni * 64 + (lane & 15);
    int koffB = (lane >> 4) * 16;
    if (threadIdx.x < TM) { bsum[threadIdx.x] = 0.f; bsq[threadIdx.x] = 0.f; }
    v8f acc[4] = {};
    int tk = threadIdx.x >> 3;
    int tn = (threadIdx.x & 7) * 16;
    const float* bsrc = Y1 + ((size_t)bz * D1C + tk) * NN + n0 + tn;

    auto stage = [&](int k0, int buf) {
        const float* s = bsrc + (size_t)k0 * NN;
        float4 f0 = ((const float4*)s)[0];
        float4 f1 = ((const float4*)s)[1];
        float4 f2 = ((const float4*)s)[2];
        float4 f3 = ((const float4*)s)[3];
        float sc = scale1[k0 + tk], sh = shift1[k0 + tk];
        float vv[16] = {f0.x, f0.y, f0.z, f0.w, f1.x, f1.y, f1.z, f1.w,
                        f2.x, f2.y, f2.z, f2.w, f3.x, f3.y, f3.z, f3.w};
#pragma unroll
        for (int i = 0; i < 16; ++i)
            ldsB[buf][tn + i][tk] = f32_to_bf16(fmaxf(fmaf(vv[i], sc, sh), 0.f));
    };
    auto compute = [&](int k0, int buf) {
        Frag af;
        const unsigned short* arow = A + (size_t)mA * D1C + k0 + koffA;
        af.q[0] = *(const uint4*)(arow);
        af.q[1] = *(const uint4*)(arow + 8);
        Frag bf0, bf1, bf2, bf3;
        const unsigned short* br0 = &ldsB[buf][nlb +  0][koffB];
        const unsigned short* br1 = &ldsB[buf][nlb + 16][koffB];
        const unsigned short* br2 = &ldsB[buf][nlb + 32][koffB];
        const unsigned short* br3 = &ldsB[buf][nlb + 48][koffB];
        bf0.q[0] = *(const uint4*)(br0); bf0.q[1] = *(const uint4*)(br0 + 8);
        bf1.q[0] = *(const uint4*)(br1); bf1.q[1] = *(const uint4*)(br1 + 8);
        bf2.q[0] = *(const uint4*)(br2); bf2.q[1] = *(const uint4*)(br2 + 8);
        bf3.q[0] = *(const uint4*)(br3); bf3.q[1] = *(const uint4*)(br3 + 8);
        acc[0] = __builtin_amdgcn_wmma_f32_16x16x32_bf16(
            false, af.v, false, bf0.v, (short)0, acc[0], false, false);
        acc[1] = __builtin_amdgcn_wmma_f32_16x16x32_bf16(
            false, af.v, false, bf1.v, (short)0, acc[1], false, false);
        acc[2] = __builtin_amdgcn_wmma_f32_16x16x32_bf16(
            false, af.v, false, bf2.v, (short)0, acc[2], false, false);
        acc[3] = __builtin_amdgcn_wmma_f32_16x16x32_bf16(
            false, af.v, false, bf3.v, (short)0, acc[3], false, false);
    };

    const int nsteps = D1C / TK;            // 8
    stage(0, 0);
    for (int t = 0; t < nsteps; ++t) {
        __syncthreads();
        if (t + 1 < nsteps) stage((t + 1) * TK, (t + 1) & 1);
        compute(t * TK, t & 1);
    }

#pragma unroll
    for (int j = 0; j < 4; ++j) {
#pragma unroll
        for (int r = 0; r < 8; ++r) {
            int mloc = mi * 16 + r + ((lane >> 4) << 3);
            int m = m0 + mloc;
            int n = n0 + ni * 64 + j * 16 + (lane & 15);
            float val = acc[j][r] + bias[m];
            Y[((size_t)bz * D2C + m) * NN + n] = val;
            float s = val, q = val * val;
#pragma unroll
            for (int off = 1; off < 16; off <<= 1) {
                s += __shfl_xor(s, off, 16);
                q += __shfl_xor(q, off, 16);
            }
            if ((lane & 15) == 0) {
                atomicAdd(&bsum[mloc], s);
                atomicAdd(&bsq[mloc], q);
            }
        }
    }
    __syncthreads();
    if (threadIdx.x < TM) {
        atomicAdd(&gsum[m0 + threadIdx.x], bsum[threadIdx.x]);
        atomicAdd(&gsq[m0 + threadIdx.x],  bsq[threadIdx.x]);
    }
}

// ---------------------------------------------------------------------------
// 8) Final BN2 + ReLU in place on d_out (float4 vectorized).
// ---------------------------------------------------------------------------
__global__ __launch_bounds__(256) void bnrelu_kernel(
    float* __restrict__ Y, const float* __restrict__ scale,
    const float* __restrict__ shift) {
    size_t i = ((size_t)blockIdx.x * 256 + threadIdx.x) * 4;
    int c = (int)((i / NN) % D2C);
    float4* p = (float4*)(Y + i);
    float4 v = *p;
    float sc = scale[c], sh = shift[c];
    v.x = fmaxf(fmaf(v.x, sc, sh), 0.f);
    v.y = fmaxf(fmaf(v.y, sc, sh), 0.f);
    v.z = fmaxf(fmaf(v.z, sc, sh), 0.f);
    v.w = fmaxf(fmaf(v.w, sc, sh), 0.f);
    *p = v;
}

// ---------------------------------------------------------------------------
extern "C" void kernel_launch(void* const* d_in, const int* in_sizes, int n_in,
                              void* d_out, int out_size, void* d_ws, size_t ws_size,
                              hipStream_t stream) {
    const float* xyz    = (const float*)d_in[0];
    const float* xyzp   = (const float*)d_in[1];
    const float* feats  = (const float*)d_in[2];
    const float* featsp = (const float*)d_in[3];
    const float* w1  = (const float*)d_in[4];
    const float* b1  = (const float*)d_in[5];
    const float* g1  = (const float*)d_in[6];
    const float* be1 = (const float*)d_in[7];
    const float* w2  = (const float*)d_in[8];
    const float* b2  = (const float*)d_in[9];
    const float* g2  = (const float*)d_in[10];
    const float* be2 = (const float*)d_in[11];
    float* out = (float*)d_out;

    char* ws = (char*)d_ws;
    int*            idx_buf = (int*)           (ws + 0);
    float*          wgt_buf = (float*)         (ws + 786432);
    unsigned short* w1b     = (unsigned short*)(ws + 1572864);
    unsigned short* w2b     = (unsigned short*)(ws + 1835008);
    float*          scale1  = (float*)         (ws + 1966080);
    float*          shift1  = (float*)         (ws + 1967104);
    float*          scale2  = (float*)         (ws + 1968128);
    float*          shift2  = (float*)         (ws + 1969152);
    float*          gsum1   = (float*)         (ws + 1970176);
    float*          gsq1    = (float*)         (ws + 1971200);
    float*          gsum2   = (float*)         (ws + 1972224);
    float*          gsq2    = (float*)         (ws + 1973248);
    unsigned short* xb      = (unsigned short*)(ws + 2097152);                 // 64 MB
    float*          y1      = (float*)         (ws + 2097152 + 67108864);     // 64 MB

    zero_kernel<<<4, 256, 0, stream>>>(gsum1);   // zeroes gsum1,gsq1,gsum2,gsq2
    knn_kernel<<<dim3(NN / 256, BB), 256, 0, stream>>>(xyz, xyzp, idx_buf, wgt_buf);
    convert_w_kernel<<<(D1C * KIN + D2C * D1C + 255) / 256, 256, 0, stream>>>(w1, w2, w1b, w2b);
    interp_concat_kernel<<<dim3(KIN, BB), 256, 0, stream>>>(feats, featsp, idx_buf, wgt_buf, xb);
    gemm1_kernel<<<dim3(NN / TN, D1C / TM, BB), 256, 0, stream>>>(w1b, xb, b1, y1, gsum1, gsq1);
    finalize_kernel<<<1, 256, 0, stream>>>(gsum1, gsq1, g1, be1, scale1, shift1);
    gemm2_kernel<<<dim3(NN / TN, D2C / TM, BB), 256, 0, stream>>>(w2b, y1, scale1, shift1, b2, out, gsum2, gsq2);
    finalize_kernel<<<1, 256, 0, stream>>>(gsum2, gsq2, g2, be2, scale2, shift2);
    bnrelu_kernel<<<((size_t)BB * D2C * NN) / (4 * 256), 256, 0, stream>>>(out, scale2, shift2);
}
